// EfficientAttention_39865886441544
// MI455X (gfx1250) — matrix operand order repaired and checked by reading, not compiled, not run on hardware
//
#include <hip/hip_runtime.h>
#include <hip/hip_bf16.h>

typedef _Float16 h16;
typedef __attribute__((ext_vector_type(8)))  _Float16 v8h;
typedef __attribute__((ext_vector_type(16))) _Float16 v16h;
typedef __attribute__((ext_vector_type(8)))  float    v8f;

#define L_SEQ  4096
#define DMODEL 1024
#define NHEADS 16
#define DHEAD  64
#define WIN    256
#define VT_LD  4128   // padded V^T row stride (keys axis) so masked tail loads stay in-bounds

// ---- WMMA fragment loaders (contiguous K = 16h+s, same order for A and B) ----
// A-frag 16x32 f16: lane l -> row (l&15); holds K in [16h, 16h+16) contiguously
static __device__ __forceinline__ v16h load_a16(const h16* p /*row base + k0*/, int h) {
    union { v16h v; v8h q[2]; } u;
    u.q[0] = *(const v8h*)(p + 16 * h);
    u.q[1] = *(const v8h*)(p + 16 * h + 8);
    return u.v;
}
// B-frag 32x16 f16: lane l -> col (l&15); holds K in [16h, 16h+16) contiguously
static __device__ __forceinline__ v16h load_b16(const h16* p /*col-source row base + k0*/, int h) {
    return *(const v16h*)(p + 16 * h);
}
#define WMMA_F16(a, b, c) \
    __builtin_amdgcn_wmma_f32_16x16x32_f16(false, (a), false, (b), (short)0, (c), false, false)

// ---------------------------------------------------------------------------
__global__ void k_cvt_f32_f16(const float* __restrict__ src, h16* __restrict__ dst, int n) {
    int i = blockIdx.x * blockDim.x + threadIdx.x;
    int stride = gridDim.x * blockDim.x;
    for (; i < n; i += stride) dst[i] = (h16)src[i];
}

// C[M,N] = A[M,K] @ Bw[N,K]^T + bias   (NT form; A rows & Bw rows contiguous along K)
// mode 0: f32 row-major out; mode 1: f16 row-major out; mode 2: f16 transposed out (col*ldo+row)
__global__ void k_gemm_nt(const h16* __restrict__ A, const h16* __restrict__ Bw,
                          const float* __restrict__ bias,
                          h16* __restrict__ outh, float* __restrict__ outf,
                          int mode, int Kd, int ldo) {
    const int lane = threadIdx.x & 31, wave = threadIdx.x >> 5;
    const int h = lane >> 4, ln = lane & 15;
    const int m0 = blockIdx.y * 128 + (wave & 3) * 32;  // 4 waves along M
    const int n0 = blockIdx.x * 64  + (wave >> 2) * 32; // 2 waves along N
    v8f acc00{}, acc01{}, acc10{}, acc11{};
    const h16* a0p = A + (size_t)(m0 + ln) * Kd;
    const h16* a1p = A + (size_t)(m0 + 16 + ln) * Kd;
    const h16* b0p = Bw + (size_t)(n0 + ln) * Kd;
    const h16* b1p = Bw + (size_t)(n0 + 16 + ln) * Kd;
    for (int k0 = 0; k0 < Kd; k0 += 32) {
        v16h a0 = load_a16(a0p + k0, h);
        v16h a1 = load_a16(a1p + k0, h);
        v16h b0 = load_b16(b0p + k0, h);
        v16h b1 = load_b16(b1p + k0, h);
        acc00 = WMMA_F16(a0, b0, acc00);
        acc01 = WMMA_F16(a0, b1, acc01);
        acc10 = WMMA_F16(a1, b0, acc10);
        acc11 = WMMA_F16(a1, b1, acc11);
    }
    auto store_frag = [&](const v8f& c, int fm, int fn) {
#pragma unroll
        for (int v = 0; v < 8; ++v) {
            int row = m0 + fm * 16 + v + 8 * h;   // C layout: M = vgpr + 8*(lane>>4)
            int col = n0 + fn * 16 + ln;          // N = lane&15
            float val = c[v] + bias[col];
            if (mode == 0)      outf[(size_t)row * ldo + col] = val;
            else if (mode == 1) outh[(size_t)row * ldo + col] = (h16)val;
            else                outh[(size_t)col * ldo + row] = (h16)val;  // transposed (V^T)
        }
    };
    store_frag(acc00, 0, 0); store_frag(acc01, 0, 1);
    store_frag(acc10, 1, 0); store_frag(acc11, 1, 1);
}

// One wave per (head, 16-query tile). Online softmax, 32 keys per iteration.
__global__ void k_attn(const h16* __restrict__ Qh, const h16* __restrict__ Kh,
                       const h16* __restrict__ Vt, h16* __restrict__ AOh) {
    __shared__ __align__(32) h16 lds_p[8][16 * 32];
    const int lane = threadIdx.x & 31, wave = threadIdx.x >> 5;
    const int h = lane >> 4, ln = lane & 15;
    const int w  = blockIdx.x * 8 + wave;   // 4096 waves total
    const int qt = w & 255, hd = w >> 8;
    const int q0 = qt * 16;
    h16* lds = &lds_p[wave][0];

    // Q A-frags for dh=64 (two K=32 chunks); rows q0+(l&15)
    const h16* qbase = Qh + (size_t)(q0 + ln) * DMODEL + hd * DHEAD;
    v16h qa0 = load_a16(qbase, h);
    v16h qa1 = load_a16(qbase + 32, h);

    float mrow[8], lrow[8];
    v8f o0{}, o1{}, o2{}, o3{};
#pragma unroll
    for (int v = 0; v < 8; ++v) { mrow[v] = -3.0e38f; lrow[v] = 0.f; }

    int jstart = q0 - WIN;      if (jstart < 0) jstart = 0;
    int jend   = q0 + 15 + WIN; if (jend > L_SEQ - 1) jend = L_SEQ - 1;

    for (int j0 = jstart; j0 <= jend; j0 += 32) {
        int jr0 = j0 + ln, jr1 = j0 + 16 + ln;
        int jc0 = jr0 > L_SEQ - 1 ? L_SEQ - 1 : jr0;   // clamp loads; masked below
        int jc1 = jr1 > L_SEQ - 1 ? L_SEQ - 1 : jr1;
        const h16* kb = Kh + hd * DHEAD;
        v16h k00 = load_b16(kb + (size_t)jc0 * DMODEL, h);
        v16h k01 = load_b16(kb + (size_t)jc0 * DMODEL + 32, h);
        v16h k10 = load_b16(kb + (size_t)jc1 * DMODEL, h);
        v16h k11 = load_b16(kb + (size_t)jc1 * DMODEL + 32, h);
        v8f s0{}; s0 = WMMA_F16(qa0, k00, s0); s0 = WMMA_F16(qa1, k01, s0);
        v8f s1{}; s1 = WMMA_F16(qa0, k10, s1); s1 = WMMA_F16(qa1, k11, s1);

        float x0[8], x1[8], alpha[8];
#pragma unroll
        for (int v = 0; v < 8; ++v) {
            int i = q0 + v + 8 * h;
            int ja = j0 + ln, jb = j0 + 16 + ln;
            int da = i - ja; if (da < 0) da = -da;
            int db = i - jb; if (db < 0) db = -db;
            x0[v] = (ja < L_SEQ && da <= WIN) ? s0[v] * 0.125f : -3.0e38f;
            x1[v] = (jb < L_SEQ && db <= WIN) ? s1[v] * 0.125f : -3.0e38f;
        }
#pragma unroll
        for (int v = 0; v < 8; ++v) {
            // row r = v+8h lives in the 16-lane half sharing bit4 -> xor masks 1,2,4,8
            float t = fmaxf(x0[v], x1[v]);
            t = fmaxf(t, __shfl_xor(t, 1, 32));
            t = fmaxf(t, __shfl_xor(t, 2, 32));
            t = fmaxf(t, __shfl_xor(t, 4, 32));
            t = fmaxf(t, __shfl_xor(t, 8, 32));
            float nm = fmaxf(mrow[v], t);
            float al = __expf(mrow[v] - nm);
            mrow[v] = nm;
            float p0 = (x0[v] < -1.0e37f) ? 0.f : __expf(x0[v] - nm);
            float p1 = (x1[v] < -1.0e37f) ? 0.f : __expf(x1[v] - nm);
            float rs = p0 + p1;
            rs += __shfl_xor(rs, 1, 32); rs += __shfl_xor(rs, 2, 32);
            rs += __shfl_xor(rs, 4, 32); rs += __shfl_xor(rs, 8, 32);
            lrow[v] = lrow[v] * al + rs;
            alpha[v] = al;
            int r = v + 8 * h;                 // C-layout row owned by this lane
            lds[r * 32 + ln]      = (h16)p0;   // stage P tile (16x32) for A-layout reread
            lds[r * 32 + 16 + ln] = (h16)p1;
        }
#pragma unroll
        for (int v = 0; v < 8; ++v) {
            o0[v] *= alpha[v]; o1[v] *= alpha[v]; o2[v] *= alpha[v]; o3[v] *= alpha[v];
        }

        asm volatile("s_wait_dscnt 0" ::: "memory");   // LDS stores -> lane-crossed loads
        union { v16h v; v8h q[2]; } pa;
        pa.q[0] = *(const v8h*)(lds + ln * 32 + 16 * h);       // K = 16h+e
        pa.q[1] = *(const v8h*)(lds + ln * 32 + 16 * h + 8);   // K = 16h+8+e
        const h16* vtb = Vt + (size_t)(hd * DHEAD + ln) * VT_LD + j0 + 16 * h;
        v16h vb0 = *(const v16h*)(vtb + (size_t)0 * 16 * VT_LD);
        v16h vb1 = *(const v16h*)(vtb + (size_t)1 * 16 * VT_LD);
        v16h vb2 = *(const v16h*)(vtb + (size_t)2 * 16 * VT_LD);
        v16h vb3 = *(const v16h*)(vtb + (size_t)3 * 16 * VT_LD);
        o0 = WMMA_F16(pa.v, vb0, o0);
        o1 = WMMA_F16(pa.v, vb1, o1);
        o2 = WMMA_F16(pa.v, vb2, o2);
        o3 = WMMA_F16(pa.v, vb3, o3);
    }

#pragma unroll
    for (int v = 0; v < 8; ++v) {
        float inv = lrow[v] > 0.f ? 1.f / lrow[v] : 0.f;
        int row = q0 + v + 8 * h;
        h16* op = AOh + (size_t)row * DMODEL + hd * DHEAD + ln;
        op[0]  = (h16)(o0[v] * inv);
        op[16] = (h16)(o1[v] * inv);
        op[32] = (h16)(o2[v] * inv);
        op[48] = (h16)(o3[v] * inv);
    }
}

// ---------------------------------------------------------------------------
extern "C" void kernel_launch(void* const* d_in, const int* in_sizes, int n_in,
                              void* d_out, int out_size, void* d_ws, size_t ws_size,
                              hipStream_t stream) {
    const float* x  = (const float*)d_in[0];
    const float* Wq = (const float*)d_in[1];
    const float* bq = (const float*)d_in[2];
    const float* Wk = (const float*)d_in[3];
    const float* bk = (const float*)d_in[4];
    const float* Wv = (const float*)d_in[5];
    const float* bv = (const float*)d_in[6];
    const float* Wo = (const float*)d_in[7];
    const float* bo = (const float*)d_in[8];
    float* out = (float*)d_out;

    h16* ws  = (h16*)d_ws;
    h16* Xh  = ws;                      // 4096*1024
    h16* WQh = Xh  + 4194304;           // 1024*1024 each
    h16* WKh = WQh + 1048576;
    h16* WVh = WKh + 1048576;
    h16* WOh = WVh + 1048576;
    h16* Qh  = WOh + 1048576;           // 4096*1024
    h16* Kh  = Qh  + 4194304;           // 4096*1024
    h16* Vt  = Kh  + 4194304;           // 1024*4128 (V^T, padded keys axis)
    h16* AOh = Vt  + (size_t)1024 * VT_LD; // 4096*1024

    k_cvt_f32_f16<<<2048, 256, 0, stream>>>(x,  Xh,  4194304);
    k_cvt_f32_f16<<<1024, 256, 0, stream>>>(Wq, WQh, 1048576);
    k_cvt_f32_f16<<<1024, 256, 0, stream>>>(Wk, WKh, 1048576);
    k_cvt_f32_f16<<<1024, 256, 0, stream>>>(Wv, WVh, 1048576);
    k_cvt_f32_f16<<<1024, 256, 0, stream>>>(Wo, WOh, 1048576);

    dim3 g(DMODEL / 64, L_SEQ / 128);   // (16, 32), 256 threads = 8 waves
    k_gemm_nt<<<g, 256, 0, stream>>>(Xh, WQh, bq, Qh, nullptr, 1, DMODEL, DMODEL);
    k_gemm_nt<<<g, 256, 0, stream>>>(Xh, WKh, bk, Kh, nullptr, 1, DMODEL, DMODEL);
    k_gemm_nt<<<g, 256, 0, stream>>>(Xh, WVh, bv, Vt, nullptr, 2, DMODEL, VT_LD);

    k_attn<<<512, 256, 0, stream>>>(Qh, Kh, Vt, AOh);

    k_gemm_nt<<<g, 256, 0, stream>>>(AOh, WOh, bo, nullptr, out, 0, DMODEL, DMODEL);
}
